// FlatQuantMLP_43207370998077
// MI455X (gfx1250) — 8-bit matrix operand order repaired and checked by reading, compiled, not run on hardware
//
#include <hip/hip_runtime.h>
#include <hip/hip_bf16.h>
#include <cstdint>

typedef signed char s8;
typedef __attribute__((ext_vector_type(8))) int   v8i;
typedef __attribute__((ext_vector_type(2))) float v2f;
typedef __attribute__((ext_vector_type(8))) float v8f;

__global__ void __launch_bounds__(256)
transform_quant_kernel(const float* __restrict__ in,
                       const float* __restrict__ Lm,
                       const float* __restrict__ Rm,
                       const float* __restrict__ diag,
                       s8* __restrict__ outq,
                       float* __restrict__ outs,
                       int l, int r, int divide)
{
    extern __shared__ float smem[];
    const int n = l * r;
    float* Xs  = smem;
    float* T   = smem + n;
    float* red = smem + 2 * n;

    const int row = blockIdx.x;
    const float* xin = in + (size_t)row * n;

    const int tid  = threadIdx.x;
    const int lane = tid & 31;
    const int wave = tid >> 5;
    const int half = lane >> 4;
    const int ml   = lane & 15;

    for (int idx = tid; idx < n; idx += 256) {
        float d = diag[idx];
        float v = xin[idx];
        Xs[idx] = divide ? (v / d) : (v * d);
    }
    __syncthreads();

    const int Mt = l >> 4;
    const int Nt = r >> 4;
    const int TT = Mt * Nt;

    for (int tile = wave; tile < TT; tile += 8) {
        const int i0 = (tile / Nt) << 4;
        const int j0 = (tile % Nt) << 4;
        v8f acc = {0.f, 0.f, 0.f, 0.f, 0.f, 0.f, 0.f, 0.f};
        for (int k = 0; k < r; k += 4) {
            v2f a, b;
#pragma unroll
            for (int v = 0; v < 2; ++v) {
                const int kg = k + half * 2 + v;
                a[v] = Xs[(i0 + ml) * r + kg];
                b[v] = Rm[(size_t)kg * r + j0 + ml];
            }
            acc = __builtin_amdgcn_wmma_f32_16x16x4_f32(
                false, a, false, b, (short)0, acc, false, false);
        }
#pragma unroll
        for (int rr = 0; rr < 8; ++rr)
            T[(i0 + rr + half * 8) * r + j0 + ml] = acc[rr];
    }
    __syncthreads();

    float amax = 0.f;
    for (int tile = wave; tile < TT; tile += 8) {
        const int i0 = (tile / Nt) << 4;
        const int j0 = (tile % Nt) << 4;
        v8f acc = {0.f, 0.f, 0.f, 0.f, 0.f, 0.f, 0.f, 0.f};
        for (int k = 0; k < l; k += 4) {
            v2f a, b;
#pragma unroll
            for (int v = 0; v < 2; ++v) {
                const int kg = k + half * 2 + v;
                a[v] = Lm[(size_t)kg * l + i0 + ml];
                b[v] = T[kg * r + j0 + ml];
            }
            acc = __builtin_amdgcn_wmma_f32_16x16x4_f32(
                false, a, false, b, (short)0, acc, false, false);
        }
#pragma unroll
        for (int rr = 0; rr < 8; ++rr) {
            float y = acc[rr];
            Xs[(i0 + rr + half * 8) * r + j0 + ml] = y;
            amax = fmaxf(amax, fabsf(y));
        }
    }

    red[tid] = amax;
    __syncthreads();
    for (int s = 128; s > 0; s >>= 1) {
        if (tid < (unsigned)s)
            red[tid] = fmaxf(red[tid], red[tid + s]);
        __syncthreads();
    }
    float scale = fmaxf(red[0] * (1.0f / 127.0f), 1e-8f);
    float inv = 1.0f / scale;
    if (tid == 0) outs[row] = scale;

    s8* orow = outq + (size_t)row * n;
    for (int idx = tid; idx < n; idx += 256) {
        float q = Xs[idx] * inv;
        q = fminf(fmaxf(q, -127.0f), 127.0f);
        orow[idx] = (s8)__float2int_rn(q);
    }
}

__device__ __forceinline__ v8i load_a_frag(const s8* __restrict__ A, int lda,
                                           int m0, int k0, int lane)
{
    int half = lane >> 4, m = lane & 15;
    const s8* base = A + (size_t)(m0 + m) * lda + k0 + half * 16;
    v8i a;
#pragma unroll
    for (int v = 0; v < 8; ++v) {
        int ks = (v & 3) * 4 + ((v >> 2) * 32);
        a[v] = *(const int*)(base + ks);
    }
    return a;
}

__device__ __forceinline__ v8i load_b_frag(const s8* __restrict__ W, int ldw,
                                           int n0, int k0, int lane)
{
    int khalf = lane >> 4, nc = lane & 15;
    const s8* base = W + (size_t)(n0 + nc) * ldw + k0 + khalf * 16;
    v8i b;
#pragma unroll
    for (int v = 0; v < 8; ++v) {
        int ks = (v & 3) * 4 + ((v >> 2) * 32);
        b[v] = *(const int*)(base + ks);
    }
    return b;
}

__global__ void __launch_bounds__(256)
gemm_upgate_swiglu(const s8* __restrict__ xq, const float* __restrict__ sx,
                   const s8* __restrict__ wu, const float* __restrict__ swu,
                   const s8* __restrict__ wg, const float* __restrict__ swg,
                   float* __restrict__ h, int M, int N, int K)
{
    int tid = threadIdx.x, lane = tid & 31, w = tid >> 5;
    int wm = w >> 2, wn = w & 3;
    int m0 = blockIdx.y * 32 + wm * 16;
    int n0 = blockIdx.x * 256 + wn * 64;

    v8i accU[4], accG[4];
#pragma unroll
    for (int t = 0; t < 4; ++t) {
        accU[t] = (v8i)(0);
        accG[t] = (v8i)(0);
    }

    for (int k0 = 0; k0 < K; k0 += 64) {
        v8i a = load_a_frag(xq, K, m0, k0, lane);
#pragma unroll
        for (int t = 0; t < 4; ++t) {
            v8i bu = load_b_frag(wu, K, n0 + t * 16, k0, lane);
            accU[t] = __builtin_amdgcn_wmma_i32_16x16x64_iu8(
                true, a, true, bu, accU[t], false, false);
            v8i bg = load_b_frag(wg, K, n0 + t * 16, k0, lane);
            accG[t] = __builtin_amdgcn_wmma_i32_16x16x64_iu8(
                true, a, true, bg, accG[t], false, false);
        }
    }

    int half = lane >> 4, nc = lane & 15;
#pragma unroll
    for (int t = 0; t < 4; ++t) {
        int n = n0 + t * 16 + nc;
        float su = swu[n], sg = swg[n];
#pragma unroll
        for (int rr = 0; rr < 8; ++rr) {
            int m = m0 + rr + half * 8;
            float sxm = sx[m];
            float u = (float)accU[t][rr] * sxm * su;
            float g = (float)accG[t][rr] * sxm * sg;
            float sig = 1.0f / (1.0f + __expf(-g));
            h[(size_t)m * N + n] = g * sig * u;
        }
    }
}

__global__ void __launch_bounds__(256)
gemm_down(const s8* __restrict__ hq, const float* __restrict__ sh,
          const s8* __restrict__ wd, const float* __restrict__ swd,
          float* __restrict__ out, int M, int N, int K)
{
    int tid = threadIdx.x, lane = tid & 31, w = tid >> 5;
    int wm = w >> 2, wn = w & 3;
    int m0 = blockIdx.y * 32 + wm * 16;
    int n0 = blockIdx.x * 256 + wn * 64;

    v8i acc[4];
#pragma unroll
    for (int t = 0; t < 4; ++t) acc[t] = (v8i)(0);

    for (int k0 = 0; k0 < K; k0 += 64) {
        v8i a = load_a_frag(hq, K, m0, k0, lane);
#pragma unroll
        for (int t = 0; t < 4; ++t) {
            v8i b = load_b_frag(wd, K, n0 + t * 16, k0, lane);
            acc[t] = __builtin_amdgcn_wmma_i32_16x16x64_iu8(
                true, a, true, b, acc[t], false, false);
        }
    }

    int half = lane >> 4, nc = lane & 15;
#pragma unroll
    for (int t = 0; t < 4; ++t) {
        int n = n0 + t * 16 + nc;
        float sn = swd[n];
#pragma unroll
        for (int rr = 0; rr < 8; ++rr) {
            int m = m0 + rr + half * 8;
            out[(size_t)m * N + n] = (float)acc[t][rr] * sh[m] * sn;
        }
    }
}

extern "C" void kernel_launch(void* const* d_in, const int* in_sizes, int n_in,
                              void* d_out, int out_size, void* d_ws, size_t ws_size,
                              hipStream_t stream)
{
    const float* x      = (const float*)d_in[0];
    const float* w_up   = (const float*)d_in[1];
    const float* w_gate = (const float*)d_in[2];
    const float* w_down = (const float*)d_in[3];
    const float* ug_L   = (const float*)d_in[4];
    const float* ug_R   = (const float*)d_in[5];
    const float* ug_d   = (const float*)d_in[6];
    const float* dn_L   = (const float*)d_in[7];
    const float* dn_R   = (const float*)d_in[8];
    const float* dn_d   = (const float*)d_in[9];
    float* out = (float*)d_out;

    const int M = 4096;
    const int HID = 4096;
    const int INTER = 14336;

    char* ws = (char*)d_ws;
    size_t off = 0;
    auto take = [&](size_t bytes) -> char* {
        char* p = ws + off;
        off = (off + bytes + 255) & ~(size_t)255;
        return p;
    };
    s8*    xq  = (s8*)   take((size_t)M * HID);
    float* sx  = (float*)take((size_t)M * 4);
    s8*    wuq = (s8*)   take((size_t)INTER * HID);
    float* swu = (float*)take((size_t)INTER * 4);
    s8*    wgq = (s8*)   take((size_t)INTER * HID);
    float* swg = (float*)take((size_t)INTER * 4);
    s8*    wdq = (s8*)   take((size_t)HID * INTER);
    float* swd = (float*)take((size_t)HID * 4);
    float* h   = (float*)take((size_t)M * INTER * 4);
    s8*    hq  = (s8*)   take((size_t)M * INTER);
    float* sh  = (float*)take((size_t)M * 4);

    const size_t sh64 = (size_t)(2 * 64 * 64 + 256) * sizeof(float);
    const size_t shdn = (size_t)(2 * 112 * 128 + 256) * sizeof(float);

    transform_quant_kernel<<<M, 256, sh64, stream>>>(
        x, ug_L, ug_R, ug_d, xq, sx, 64, 64,  0);
    transform_quant_kernel<<<INTER, 256, sh64, stream>>>(
        w_up, ug_L, ug_R, ug_d, wuq, swu, 64, 64,  1);
    transform_quant_kernel<<<INTER, 256, sh64, stream>>>(
        w_gate, ug_L, ug_R, ug_d, wgq, swg, 64, 64,  1);
    transform_quant_kernel<<<HID, 256, shdn, stream>>>(
        w_down, dn_L, dn_R, dn_d, wdq, swd, 112, 128,  1);

    gemm_upgate_swiglu<<<dim3(INTER / 256, M / 32), 256, 0, stream>>>(
        xq, sx, wuq, swu, wgq, swg, h, M, INTER, HID);

    transform_quant_kernel<<<M, 256, shdn, stream>>>(
        h, dn_L, dn_R, dn_d, hq, sh, 112, 128,  0);

    gemm_down<<<dim3(HID / 256, M / 32), 256, 0, stream>>>(
        hq, sh, wdq, swd, out, M, HID, INTER);
}
